// IdiomExtractor_45672682226577
// MI455X (gfx1250) — compile-verified, matrix operand order checked by reading, not run on hardware
//
#include <hip/hip_runtime.h>
#include <hip/hip_bf16.h>
#include <math.h>

// ---------------- problem constants ----------------
constexpr int Bsz = 64;
constexpr int Ssz = 512;
constexpr int H   = 768;
constexpr int HD  = 384;            // hidden per direction
constexpr int G   = 4 * HD;         // 1536 gates
constexpr int Ccl = 5;
constexpr int ROWS = Bsz * Ssz;     // 32768

typedef __attribute__((ext_vector_type(16))) __bf16       v16bf;
typedef __attribute__((ext_vector_type(8)))  float        v8f;
typedef __attribute__((ext_vector_type(4)))  unsigned int u32x4;
typedef __attribute__((ext_vector_type(8)))  unsigned int u32x8;

// ---------------- WMMA fragment helpers (wave32 layouts per CDNA5 ISA §7.12.2) ----
__device__ __forceinline__ v16bf load_frag(const __bf16* p, int ld, int rowbase,
                                           int k0, int lane) {
  int r  = rowbase + (lane & 15);
  int kb = k0 + ((lane >> 4) << 3);
  const __bf16* q = p + (size_t)r * ld + kb;
  union { v16bf v; __bf16 e[16]; } f;
#pragma unroll
  for (int j = 0; j < 8; ++j) { f.e[j] = q[j]; f.e[j + 8] = q[j + 16]; }
  return f.v;
}

__device__ __forceinline__ void store_tile(float* out, int ld, int mbase, int nbase,
                                           int lane, const v8f& c) {
  int n  = nbase + (lane & 15);
  int mo = mbase + ((lane >> 4) << 3);
#pragma unroll
  for (int r = 0; r < 8; ++r) out[(size_t)(mo + r) * ld + n] = c[r];
}

__device__ __forceinline__ v8f wmma_bf16(const v16bf& a, const v16bf& b, const v8f& c) {
  return __builtin_amdgcn_wmma_f32_16x16x32_bf16(false, a, false, b, (short)0, c,
                                                 false, false);
}

__device__ __forceinline__ float sigmoidf_(float x) { return 1.0f / (1.0f + expf(-x)); }

// ---------------- TDM: 2D tile Global -> LDS (CDNA5 ISA §7/§8, async_tensor) -----
// D# group0 (128b): [1:0]=count=1, [63:32]=lds_addr, [120:64]=global_addr,
//                   [127:126]=type=2.
// D# group1 (256b): [17:16]=data_size (1 -> 2 bytes), [79:48]=tensor_dim0,
//                   [111:80]=tensor_dim1, [127:112]=tile_dim0, [143:128]=tile_dim1,
//                   [207:160]=tensor_dim0_stride; tile_dim2 / dim1_stride = 0 (2D).
__device__ __forceinline__ void tdm_load_2d_bf16(unsigned int lds_off, const void* gptr,
                                                 unsigned int elems_per_row,
                                                 unsigned int rows,
                                                 unsigned int row_stride_elems) {
  unsigned long long ga = (unsigned long long)(size_t)gptr;
  u32x4 g0;
  g0[0] = 1u;                                               // count = 1
  g0[1] = lds_off;                                          // lds_addr (bytes)
  g0[2] = (unsigned int)ga;                                 // global_addr lo32
  g0[3] = (unsigned int)((ga >> 32) & 0x1FFFFFFu) | 0x80000000u;  // hi25 | type=2
  u32x8 g1;
  g1[0] = (1u << 16);                                       // data_size = 2 bytes
  g1[1] = (elems_per_row & 0xFFFFu) << 16;                  // tensor_dim0 lo16
  g1[2] = ((elems_per_row >> 16) & 0xFFFFu) | ((rows & 0xFFFFu) << 16);
  g1[3] = ((rows >> 16) & 0xFFFFu) | ((elems_per_row & 0xFFFFu) << 16); // tile_dim0
  g1[4] = rows & 0xFFFFu;                                   // tile_dim1 (tile_dim2=0)
  g1[5] = row_stride_elems;                                 // dim0_stride lo32
  g1[6] = 0u;                                               // dim0_stride hi16, dim1_stride lo16
  g1[7] = 0u;                                               // dim1_stride hi32
  asm volatile("tensor_load_to_lds %0, %1" :: "s"(g0), "s"(g1) : "memory");
}

// ---------------- utility kernels ----------------
__global__ void k_f32_to_bf16(const float* __restrict__ src, __bf16* __restrict__ dst,
                              size_t n) {
  size_t i = (size_t)blockIdx.x * blockDim.x + threadIdx.x;
  if (i < n) dst[i] = (__bf16)src[i];
}

__global__ void k_zero32(unsigned int* __restrict__ p, size_t n) {
  size_t i = (size_t)blockIdx.x * blockDim.x + threadIdx.x;
  if (i < n) p[i] = 0u;
}

// ---------------- K1: xg = x @ w_ih^T + (b_ih + b_hh), both directions ----------
__global__ void k_in_proj(const __bf16* __restrict__ x, const __bf16* __restrict__ wih,
                          const float* __restrict__ bih_f, const float* __restrict__ bhh_f,
                          const float* __restrict__ bih_b, const float* __restrict__ bhh_b,
                          float* __restrict__ xg) {
  int lane = threadIdx.x & 31, wave = threadIdx.x >> 5;
  int dir  = blockIdx.z;
  const __bf16* w  = wih + (size_t)dir * G * H;
  float*        o  = xg  + (size_t)dir * ROWS * G;
  const float*  bi = dir ? bih_b : bih_f;
  const float*  bh = dir ? bhh_b : bhh_f;

  int mbase  = blockIdx.y * 64 + (wave & 3) * 16;
  int nbase0 = blockIdx.x * 128 + (wave >> 2) * 64;

  v8f acc[4];
#pragma unroll
  for (int i = 0; i < 4; ++i)
#pragma unroll
    for (int r = 0; r < 8; ++r) acc[i][r] = 0.0f;

  for (int k0 = 0; k0 < H; k0 += 32) {
    // prefetch next A chunk (global_prefetch_b8)
    if (k0 + 64 < H)
      __builtin_prefetch(x + (size_t)(mbase + (lane & 15)) * H + k0 + 64, 0, 0);
    v16bf a = load_frag(x, H, mbase, k0, lane);
#pragma unroll
    for (int i = 0; i < 4; ++i) {
      v16bf b = load_frag(w, H, nbase0 + 16 * i, k0, lane);
      acc[i] = wmma_bf16(a, b, acc[i]);
    }
  }
#pragma unroll
  for (int i = 0; i < 4; ++i) {
    int n = nbase0 + 16 * i + (lane & 15);
    float bias = bi[n] + bh[n];
#pragma unroll
    for (int r = 0; r < 8; ++r) acc[i][r] += bias;
    store_tile(o, G, mbase, nbase0 + 16 * i, lane, acc[i]);
  }
}

// ---------------- K2: one LSTM timestep, both directions ----------------
// grid: (HD/32, 1, 2), block 256 (8 waves).
// Wave 0 stages the block's w_hh slice (4 gate chunks x 32 rows x 384 K) and the
// full h_prev (64 x 384) into LDS with TDM (tensor_load_to_lds), waits TENSORcnt,
// then all waves run the WMMA K-loop out of LDS.
__global__ void k_lstm_step(const float* __restrict__ xg, const __bf16* __restrict__ whh,
                            const __bf16* __restrict__ h_prev, __bf16* __restrict__ h_next,
                            float* __restrict__ c_state, float* __restrict__ h_seq,
                            int t_fwd) {
  extern __shared__ char smem[];
  __bf16* w_lds = (__bf16*)smem;                       // 4*32*384 bf16 = 96 KB
  __bf16* h_lds = (__bf16*)(smem + 4 * 32 * HD * 2);   // 64*384 bf16  = 48 KB

  int lane = threadIdx.x & 31, wave = threadIdx.x >> 5;
  int dir  = blockIdx.z;
  int t    = dir ? (Ssz - 1 - t_fwd) : t_fwd;

  const float*  xgd = xg     + (size_t)dir * ROWS * G;
  const __bf16* w   = whh    + (size_t)dir * G * HD;
  const __bf16* hp  = h_prev + (size_t)dir * Bsz * HD;
  __bf16*       hn  = h_next + (size_t)dir * Bsz * HD;
  float*        cs  = c_state+ (size_t)dir * Bsz * HD;
  float*        hs  = h_seq  + (size_t)dir * ROWS * HD;

  int jblk = blockIdx.x * 32;                          // 32 j-columns per block

  if (wave == 0) {
    unsigned int lds0 = (unsigned int)(size_t)smem;
#pragma unroll
    for (int q = 0; q < 4; ++q)
      tdm_load_2d_bf16(lds0 + (unsigned)(q * 32 * HD * 2),
                       w + (size_t)(q * HD + jblk) * HD, HD, 32, HD);
    tdm_load_2d_bf16(lds0 + (unsigned)(4 * 32 * HD * 2), hp, HD, Bsz, HD);
    __builtin_amdgcn_s_wait_tensorcnt(0);
  }
  __syncthreads();

  int mbase = (wave & 3) * 16;
  int jloc  = (wave >> 2) * 16;                        // local j-tile inside 32-slice

  v8f acc[4];
#pragma unroll
  for (int q = 0; q < 4; ++q)
#pragma unroll
    for (int r = 0; r < 8; ++r) acc[q][r] = 0.0f;

  for (int k0 = 0; k0 < HD; k0 += 32) {
    v16bf a = load_frag(h_lds, HD, mbase, k0, lane);
#pragma unroll
    for (int q = 0; q < 4; ++q) {
      v16bf b = load_frag(w_lds, HD, q * 32 + jloc, k0, lane);
      acc[q] = wmma_bf16(a, b, acc[q]);
    }
  }

  int j = jblk + jloc + (lane & 15);
#pragma unroll
  for (int r = 0; r < 8; ++r) {
    int m = mbase + r + ((lane >> 4) << 3);            // batch index
    const float* xr = xgd + ((size_t)m * Ssz + t) * G;
    float gi = acc[0][r] + xr[0 * HD + j];
    float gf = acc[1][r] + xr[1 * HD + j];
    float gg = acc[2][r] + xr[2 * HD + j];
    float go = acc[3][r] + xr[3 * HD + j];
    size_t idx = (size_t)m * HD + j;
    float c = sigmoidf_(gf) * cs[idx] + sigmoidf_(gi) * tanhf(gg);
    float h = sigmoidf_(go) * tanhf(c);
    cs[idx] = c;
    hn[idx] = (__bf16)h;
    hs[((size_t)m * Ssz + t) * HD + j] = h;
  }
}

// ---------------- K3: layer-norm of concat(fwd,bwd), output bf16 ------------
__global__ void k_ln_concat(const float* __restrict__ hf, const float* __restrict__ hb,
                            __bf16* __restrict__ xln) {
  int lane = threadIdx.x & 31, wave = threadIdx.x >> 5;
  int row  = blockIdx.x * 8 + wave;
  const float* pf = hf + (size_t)row * HD;
  const float* pb = hb + (size_t)row * HD;
  float s = 0.f, s2 = 0.f;
  for (int c = lane; c < H; c += 32) {
    float v = (c < HD) ? pf[c] : pb[c - HD];
    s += v; s2 += v * v;
  }
#pragma unroll
  for (int off = 16; off > 0; off >>= 1) {
    s  += __shfl_xor(s,  off);
    s2 += __shfl_xor(s2, off);
  }
  float mu = s / H;
  float rs = rsqrtf(s2 / H - mu * mu + 1e-5f);
  for (int c = lane; c < H; c += 32) {
    float v = (c < HD) ? pf[c] : pb[c - HD];
    xln[(size_t)row * H + c] = (__bf16)((v - mu) * rs);
  }
}

// ---------------- K4: h1 = relu(LN(xln @ w1^T + b1; g1,beta1)) ------------
__global__ void k_mlp1(const __bf16* __restrict__ xln, const __bf16* __restrict__ w1,
                       const float* __restrict__ b1, const float* __restrict__ g1,
                       const float* __restrict__ be1, __bf16* __restrict__ h1) {
  __shared__ float sm[16 * 256];
  __shared__ float ps[16][17], ps2[16][17];
  __shared__ float mu_s[16], rs_s[16];
  int lane = threadIdx.x & 31, wave = threadIdx.x >> 5;
  int mbase = blockIdx.x * 16;

  v8f a0, a1;
#pragma unroll
  for (int r = 0; r < 8; ++r) { a0[r] = 0.f; a1[r] = 0.f; }
  for (int k0 = 0; k0 < H; k0 += 32) {
    v16bf a  = load_frag(xln, H, mbase, k0, lane);
    v16bf b0 = load_frag(w1, H, wave * 32,      k0, lane);
    v16bf b1f= load_frag(w1, H, wave * 32 + 16, k0, lane);
    a0 = wmma_bf16(a, b0,  a0);
    a1 = wmma_bf16(a, b1f, a1);
  }
  store_tile(sm, 256, 0, wave * 32,      lane, a0);
  store_tile(sm, 256, 0, wave * 32 + 16, lane, a1);
  __syncthreads();

  int row = threadIdx.x >> 4, seg = threadIdx.x & 15;
  float s = 0.f, s2 = 0.f;
  for (int c = seg * 16; c < seg * 16 + 16; ++c) {
    float v = sm[row * 256 + c] + b1[c];
    s += v; s2 += v * v;
  }
  ps[row][seg] = s; ps2[row][seg] = s2;
  __syncthreads();
  if (seg == 0) {
    float a = 0.f, a2 = 0.f;
    for (int i = 0; i < 16; ++i) { a += ps[row][i]; a2 += ps2[row][i]; }
    float mu = a / 256.f;
    mu_s[row] = mu;
    rs_s[row] = rsqrtf(a2 / 256.f - mu * mu + 1e-5f);
  }
  __syncthreads();
  for (int idx = threadIdx.x; idx < 16 * 256; idx += 256) {
    int r = idx >> 8, c = idx & 255;
    float v = sm[idx] + b1[c];
    float y = (v - mu_s[r]) * rs_s[r] * g1[c] + be1[c];
    h1[(size_t)(mbase + r) * 256 + c] = (__bf16)fmaxf(y, 0.f);
  }
}

// ---------------- K5: h2 = relu(LN(h1 @ w2^T + b2; g2,beta2)), f32 out -------
__global__ void k_mlp2(const __bf16* __restrict__ h1, const __bf16* __restrict__ w2,
                       const float* __restrict__ b2, const float* __restrict__ g2,
                       const float* __restrict__ be2, float* __restrict__ h2) {
  __shared__ float sm[16 * 128];
  __shared__ float ps[16][17], ps2[16][17];
  __shared__ float mu_s[16], rs_s[16];
  int lane = threadIdx.x & 31, wave = threadIdx.x >> 5;
  int mbase = blockIdx.x * 16;

  v8f a0;
#pragma unroll
  for (int r = 0; r < 8; ++r) a0[r] = 0.f;
  for (int k0 = 0; k0 < 256; k0 += 32) {
    v16bf a = load_frag(h1, 256, mbase, k0, lane);
    v16bf b = load_frag(w2, 256, wave * 16, k0, lane);
    a0 = wmma_bf16(a, b, a0);
  }
  store_tile(sm, 128, 0, wave * 16, lane, a0);
  __syncthreads();

  int row = threadIdx.x >> 4, seg = threadIdx.x & 15;
  float s = 0.f, s2 = 0.f;
  for (int c = seg * 8; c < seg * 8 + 8; ++c) {
    float v = sm[row * 128 + c] + b2[c];
    s += v; s2 += v * v;
  }
  ps[row][seg] = s; ps2[row][seg] = s2;
  __syncthreads();
  if (seg == 0) {
    float a = 0.f, a2 = 0.f;
    for (int i = 0; i < 16; ++i) { a += ps[row][i]; a2 += ps2[row][i]; }
    float mu = a / 128.f;
    mu_s[row] = mu;
    rs_s[row] = rsqrtf(a2 / 128.f - mu * mu + 1e-5f);
  }
  __syncthreads();
  for (int idx = threadIdx.x; idx < 16 * 128; idx += 256) {
    int r = idx >> 7, c = idx & 127;
    float v = sm[idx] + b2[c];
    float y = (v - mu_s[r]) * rs_s[r] * g2[c] + be2[c];
    h2[(size_t)(mbase + r) * 128 + c] = fmaxf(y, 0.f);
  }
}

// ---------------- K6: emissions = h2 @ w3^T + b3 (tiny N=5, scalar f32) ------
__global__ void k_emis(const float* __restrict__ h2, const float* __restrict__ w3,
                       const float* __restrict__ b3, float* __restrict__ em) {
  int tid = blockIdx.x * blockDim.x + threadIdx.x;
  if (tid >= ROWS * Ccl) return;
  int row = tid / Ccl, c = tid % Ccl;
  const float* hr = h2 + (size_t)row * 128;
  const float* wr = w3 + c * 128;
  float s = b3[c];
  for (int k = 0; k < 128; ++k) s += hr[k] * wr[k];
  em[tid] = s;
}

// ---------------- K7: CRF log-likelihood (one thread per batch row) ----------
__global__ void k_crf(const float* __restrict__ em, const int* __restrict__ labels,
                      const float* __restrict__ start, const float* __restrict__ endw,
                      const float* __restrict__ trans, float* __restrict__ out_llh) {
  __shared__ float red[64];
  int b = threadIdx.x;
  float tr[Ccl][Ccl];
#pragma unroll
  for (int i = 0; i < Ccl; ++i)
#pragma unroll
    for (int j = 0; j < Ccl; ++j) tr[i][j] = trans[i * Ccl + j];

  const float* emb = em + (size_t)b * Ssz * Ccl;
  const int*   lb  = labels + b * Ssz;

  int prev = lb[0];
  float num = start[prev] + emb[prev];
  int mcnt = 1;
  for (int t = 1; t < Ssz; ++t) {
    int tag = lb[t];
    if (tag != 0) { num += tr[prev][tag] + emb[t * Ccl + tag]; mcnt++; }
    prev = tag;
  }
  num += endw[lb[mcnt - 1]];

  float alpha[Ccl];
#pragma unroll
  for (int j = 0; j < Ccl; ++j) alpha[j] = start[j] + emb[j];
  for (int t = 1; t < Ssz; ++t) {
    float na[Ccl];
#pragma unroll
    for (int j = 0; j < Ccl; ++j) {
      float mx = -1e30f;
#pragma unroll
      for (int i = 0; i < Ccl; ++i) mx = fmaxf(mx, alpha[i] + tr[i][j]);
      float sum = 0.f;
#pragma unroll
      for (int i = 0; i < Ccl; ++i) sum += expf(alpha[i] + tr[i][j] - mx);
      na[j] = mx + logf(sum) + emb[t * Ccl + j];
    }
    if (lb[t] != 0) {
#pragma unroll
      for (int j = 0; j < Ccl; ++j) alpha[j] = na[j];
    }
  }
  float mx = -1e30f;
#pragma unroll
  for (int j = 0; j < Ccl; ++j) mx = fmaxf(mx, alpha[j] + endw[j]);
  float sum = 0.f;
#pragma unroll
  for (int j = 0; j < Ccl; ++j) sum += expf(alpha[j] + endw[j] - mx);
  float den = mx + logf(sum);

  red[b] = num - den;
  __syncthreads();
  if (b == 0) {
    float t = 0.f;
    for (int i = 0; i < 64; ++i) t += red[i];
    out_llh[0] = t;
  }
}

// ---------------- host-side orchestration ----------------
static inline size_t align256(size_t x) { return (x + 255) & ~(size_t)255; }

extern "C" void kernel_launch(void* const* d_in, const int* in_sizes, int n_in,
                              void* d_out, int out_size, void* d_ws, size_t ws_size,
                              hipStream_t stream) {
  (void)in_sizes; (void)n_in; (void)out_size; (void)ws_size;
  const float* x_f     = (const float*)d_in[0];
  const int*   labels  = (const int*)d_in[1];
  const float* w_ih_f  = (const float*)d_in[2];
  const float* w_hh_f  = (const float*)d_in[3];
  const float* b_ih_f  = (const float*)d_in[4];
  const float* b_hh_f  = (const float*)d_in[5];
  const float* w_ih_b  = (const float*)d_in[6];
  const float* w_hh_b  = (const float*)d_in[7];
  const float* b_ih_b  = (const float*)d_in[8];
  const float* b_hh_b  = (const float*)d_in[9];
  const float* w1      = (const float*)d_in[10];
  const float* b1      = (const float*)d_in[11];
  const float* g1      = (const float*)d_in[12];
  const float* be1     = (const float*)d_in[13];
  const float* w2      = (const float*)d_in[14];
  const float* b2      = (const float*)d_in[15];
  const float* g2      = (const float*)d_in[16];
  const float* be2     = (const float*)d_in[17];
  const float* w3      = (const float*)d_in[18];
  const float* b3      = (const float*)d_in[19];
  const float* crf_s   = (const float*)d_in[20];
  const float* crf_e   = (const float*)d_in[21];
  const float* crf_t   = (const float*)d_in[22];

  float* out = (float*)d_out;            // out[0] = llh, out[1..] = emissions

  char* ws = (char*)d_ws;
  size_t off = 0;
  auto take = [&](size_t bytes) { char* p = ws + off; off += align256(bytes); return p; };

  __bf16* x_bf   = (__bf16*)take((size_t)ROWS * H * 2);
  __bf16* wih_bf = (__bf16*)take((size_t)2 * G * H * 2);
  __bf16* whh_bf = (__bf16*)take((size_t)2 * G * HD * 2);
  __bf16* w1_bf  = (__bf16*)take((size_t)256 * H * 2);
  __bf16* w2_bf  = (__bf16*)take((size_t)128 * 256 * 2);
  float*  xg     = (float*)take((size_t)2 * ROWS * G * 4);
  float*  h_seq  = (float*)take((size_t)2 * ROWS * HD * 4);
  __bf16* h_st   = (__bf16*)take((size_t)2 * 2 * Bsz * HD * 2);  // [buf][dir][B][HD]
  float*  c_st   = (float*)take((size_t)2 * Bsz * HD * 4);
  __bf16* xln    = (__bf16*)take((size_t)ROWS * H * 2);
  __bf16* h1     = (__bf16*)take((size_t)ROWS * 256 * 2);
  float*  h2     = (float*)take((size_t)ROWS * 128 * 4);

  auto cvt = [&](const float* s, __bf16* d, size_t n) {
    k_f32_to_bf16<<<(unsigned)((n + 255) / 256), 256, 0, stream>>>(s, d, n);
  };
  cvt(x_f, x_bf, (size_t)ROWS * H);
  cvt(w_ih_f, wih_bf,               (size_t)G * H);
  cvt(w_ih_b, wih_bf + (size_t)G*H, (size_t)G * H);
  cvt(w_hh_f, whh_bf,                (size_t)G * HD);
  cvt(w_hh_b, whh_bf + (size_t)G*HD, (size_t)G * HD);
  cvt(w1, w1_bf, (size_t)256 * H);
  cvt(w2, w2_bf, (size_t)128 * 256);

  {
    size_t nh = (size_t)2 * 2 * Bsz * HD * 2 / 4;
    k_zero32<<<(unsigned)((nh + 255) / 256), 256, 0, stream>>>((unsigned int*)h_st, nh);
    size_t nc = (size_t)2 * Bsz * HD;
    k_zero32<<<(unsigned)((nc + 255) / 256), 256, 0, stream>>>((unsigned int*)c_st, nc);
  }

  // K1: input projections, both directions
  k_in_proj<<<dim3(G / 128, ROWS / 64, 2), 256, 0, stream>>>(
      x_bf, wih_bf, b_ih_f, b_hh_f, b_ih_b, b_hh_b, xg);

  // K2: 512 sequential timesteps (ping-pong hidden state), TDM-staged LDS operands
  const size_t HSTRIDE = (size_t)2 * Bsz * HD;
  const unsigned lds_bytes = 4 * 32 * HD * 2 + Bsz * HD * 2;   // 96KB + 48KB
  for (int t = 0; t < Ssz; ++t) {
    __bf16* hp = h_st + (size_t)(t & 1) * HSTRIDE;
    __bf16* hn = h_st + (size_t)((t + 1) & 1) * HSTRIDE;
    k_lstm_step<<<dim3(HD / 32, 1, 2), 256, lds_bytes, stream>>>(xg, whh_bf, hp, hn,
                                                                 c_st, h_seq, t);
  }

  // K3: LN(concat(fwd,bwd)) -> bf16
  k_ln_concat<<<ROWS / 8, 256, 0, stream>>>(h_seq, h_seq + (size_t)ROWS * HD, xln);

  // K4 / K5: MLP with fused LN+ReLU
  k_mlp1<<<ROWS / 16, 256, 0, stream>>>(xln, w1_bf, b1, g1, be1, h1);
  k_mlp2<<<ROWS / 16, 256, 0, stream>>>(h1, w2_bf, b2, g2, be2, h2);

  // K6: emissions straight into d_out[1..]
  k_emis<<<(ROWS * Ccl + 255) / 256, 256, 0, stream>>>(h2, w3, b3, out + 1);

  // K7: CRF log-likelihood into d_out[0]
  k_crf<<<1, 64, 0, stream>>>(out + 1, labels, crf_s, crf_e, crf_t, out);
}
